// PointNet2DenseEncoder_8289286881649
// MI455X (gfx1250) — matrix operand order repaired and checked by reading, not compiled, not run on hardware
//
#include <hip/hip_runtime.h>

typedef __attribute__((ext_vector_type(16))) _Float16 v16h;
typedef __attribute__((ext_vector_type(8)))  _Float16 v8h;
typedef __attribute__((ext_vector_type(8)))  float    v8f;

#define BATCH 16
#define NPTS1 16384
#define S1    512
#define NS1   32
#define S2    128
#define NS2   64
#define R1    0.04f
#define R2    0.08f

// ---- workspace layout (bytes) ----
#define OFF_FPS1   0u          // int  [16*512]
#define OFF_L1XYZ  32768u      // f32  [16*512*3]
#define OFF_BALL1  131072u     // int  [16*512*32]
#define OFF_L1PTS  1179648u    // f32  [16*512*128]
#define OFF_FPS2   5373952u    // int  [16*128]
#define OFF_L2XYZ  5382144u    // f32  [16*128*3]
#define OFF_BALL2  5406720u    // int  [16*128*64]
#define OFF_W      5931008u    // f16  [131072]
// weight sub-offsets (halves). Weights kept in natural [N][K] row-major (W = [out][in]),
// so a 16-row N-slice is a contiguous 16*K-half block (async-copy friendly, B-frag = 1 vector load).
#define WO_SA1W2   0       // [64][64]
#define WO_SA1W3   4096    // [128][64]
#define WO_SA2W1   12288   // [128][160]  (K padded 131->160, zero fill)
#define WO_SA2W2   32768   // [256][128]
#define WO_SA2W3   65536   // [256][256]
#define W_TOTAL    131072

// ---------------- LDS address helper (true addrspace(3) offset for async-copy VDST) ----------------
__device__ __forceinline__ unsigned lds_addr32(const void* p) {
  return (unsigned)(unsigned long long)(__attribute__((address_space(3))) const void*)p;
}

// ---------------- async global->LDS staging (GLOBAL_LOAD_ASYNC_TO_LDS_B128, ASYNCcnt) -------------
__device__ __forceinline__ void stage_async(const _Float16* __restrict__ gsrc,
                                            _Float16* __restrict__ ldst,
                                            int halves, int tid, int nthreads) {
  const int chunks = halves >> 3;  // 16-byte chunks
  for (int i = tid; i < chunks; i += nthreads) {
    const unsigned           l  = lds_addr32(ldst + i * 8);
    const unsigned long long g  = (unsigned long long)(const void*)(gsrc + i * 8);
    asm volatile("global_load_async_to_lds_b128 %0, %1, off" :: "v"(l), "v"(g) : "memory");
  }
#if __has_builtin(__builtin_amdgcn_s_wait_asynccnt)
  __builtin_amdgcn_s_wait_asynccnt(0);
#else
  asm volatile("s_wait_asynccnt 0" ::: "memory");
#endif
}

// ---------------- WMMA fragment loaders (contiguous k = 16h + s order for both A and B, wave32) ----------------
// A: 16x32 f16. lane m=lane&15 selects row; lanes 0-15 K=kBase+0..15, lanes 16-31 K=kBase+16..31.
// 16 contiguous halves -> one aligned 32B LDS load.
__device__ __forceinline__ v16h load_a_frag(const _Float16* __restrict__ src,
                                            int rowBase, int kBase, int stride) {
  const int lane = threadIdx.x & 31;
  const int kb   = kBase + ((lane >> 4) << 4);
  const _Float16* row = src + (rowBase + (lane & 15)) * stride + kb;
  return *(const v16h*)(row);
}

// B: slice stored [16][K] (N-major == natural W layout). lanes 0-15: N=lane, K=kBase+0..15;
// lanes 16-31: N=lane-16, K=kBase+16..31. 16 contiguous halves -> one aligned 32B LDS load.
__device__ __forceinline__ v16h load_b_frag(const _Float16* __restrict__ slice, int kBase, int K) {
  const int lane = threadIdx.x & 31;
  const int kb   = kBase + ((lane >> 4) << 4);
  return *(const v16h*)(slice + (lane & 15) * K + kb);
}

// ---------------- weight prep: f32 -> f16, natural [N][K] order (pad sa2_w1 K to 160) ------------
__global__ void prep_weights_kernel(const float* __restrict__ w2a, const float* __restrict__ w3a,
                                    const float* __restrict__ w1b, const float* __restrict__ w2b,
                                    const float* __restrict__ w3b, _Float16* __restrict__ dst) {
  const int i = blockIdx.x * blockDim.x + threadIdx.x;
  if (i >= W_TOTAL) return;
  float v;
  if (i < 4096)        { v = w2a[i]; }
  else if (i < 12288)  { v = w3a[i - 4096]; }
  else if (i < 32768)  { int j = i - 12288; int n = j / 160, k = j % 160;
                         v = (k < 131) ? w1b[n * 131 + k] : 0.0f; }
  else if (i < 65536)  { v = w2b[i - 32768]; }
  else                 { v = w3b[i - 65536]; }
  dst[i] = (_Float16)v;
}

// ---------------- farthest point sampling: one block per batch ----------------
template <int NPTS, int NPOINT>
__global__ void fps_kernel(const float* __restrict__ xyz, int* __restrict__ fidx,
                           float* __restrict__ nxyz) {
  const int b = blockIdx.x;
  const int t = threadIdx.x;
  constexpr int PT = NPTS / 256;
  float dist[PT];
#pragma unroll
  for (int i = 0; i < PT; ++i) dist[i] = 1e10f;

  __shared__ float sval[256];
  __shared__ int   sidx[256];
  __shared__ int   s_far;
  __shared__ float scen[3];
  if (t == 0) s_far = 0;
  __syncthreads();

  const float* bx = xyz + (size_t)b * NPTS * 3;
  for (int it = 0; it < NPOINT; ++it) {
    if (t == 0) {
      const int f = s_far;
      fidx[b * NPOINT + it] = f;
      const float x = bx[f * 3 + 0], y = bx[f * 3 + 1], z = bx[f * 3 + 2];
      scen[0] = x; scen[1] = y; scen[2] = z;
      nxyz[(size_t)(b * NPOINT + it) * 3 + 0] = x;
      nxyz[(size_t)(b * NPOINT + it) * 3 + 1] = y;
      nxyz[(size_t)(b * NPOINT + it) * 3 + 2] = z;
    }
    __syncthreads();
    const float cx = scen[0], cy = scen[1], cz = scen[2];
    float best = -1.0f;
    int   bi   = 0;
#pragma unroll
    for (int i = 0; i < PT; ++i) {
      const int j = i * 256 + t;
      const float dx = bx[j * 3 + 0] - cx;
      const float dy = bx[j * 3 + 1] - cy;
      const float dz = bx[j * 3 + 2] - cz;
      const float d  = dx * dx + dy * dy + dz * dz;
      dist[i] = fminf(dist[i], d);
      if (dist[i] > best) { best = dist[i]; bi = j; }
    }
    sval[t] = best; sidx[t] = bi;
    __syncthreads();
    for (int off = 128; off > 0; off >>= 1) {
      if (t < off && sval[t + off] > sval[t]) { sval[t] = sval[t + off]; sidx[t] = sidx[t + off]; }
      __syncthreads();
    }
    if (t == 0) s_far = sidx[0];
    __syncthreads();
  }
}

// ---------------- ball query: one wave32 per center, ordered append via ballot ----------------
template <int NPTS, int S, int NSAMPLE>
__global__ void ball_kernel(const float* __restrict__ xyz, const float* __restrict__ nxyz,
                            int* __restrict__ ball, float r2) {
  const int w    = threadIdx.x >> 5;
  const int lane = threadIdx.x & 31;
  const int cid  = blockIdx.x * 8 + w;
  const int b    = cid / S;
  const int s    = cid % S;
  __shared__ int sbuf[8][NSAMPLE];

  const float* bx = xyz + (size_t)b * NPTS * 3;
  const float cx = nxyz[(size_t)(b * S + s) * 3 + 0];
  const float cy = nxyz[(size_t)(b * S + s) * 3 + 1];
  const float cz = nxyz[(size_t)(b * S + s) * 3 + 2];

  int cnt = 0;
  for (int chunk = 0; chunk < NPTS / 32; ++chunk) {
    const int j = chunk * 32 + lane;
    const float dx = bx[j * 3 + 0] - cx;
    const float dy = bx[j * 3 + 1] - cy;
    const float dz = bx[j * 3 + 2] - cz;
    const bool inside = (dx * dx + dy * dy + dz * dz) <= r2;
    const unsigned long long mask = __ballot(inside);
    if (inside) {
      const int pos = cnt + __popcll(mask & ((1ull << lane) - 1ull));
      if (pos < NSAMPLE) sbuf[w][pos] = j;
    }
    cnt += __popcll(mask);
    if (cnt >= NSAMPLE) break;   // uniform across the wave
  }
  __syncthreads();
  const int c     = cnt < NSAMPLE ? cnt : NSAMPLE;
  const int first = (c > 0) ? sbuf[w][0] : 0;
  for (int k = lane; k < NSAMPLE; k += 32)
    ball[(size_t)(b * S + s) * NSAMPLE + k] = (k < c) ? sbuf[w][k] : first;
}

// ---------------- SA1 MLP (3->64->64->128) + maxpool over 32 : 4 centers/block ----------------
__global__ void __launch_bounds__(256) sa1_mlp_kernel(
    const float* __restrict__ xyz, const float* __restrict__ nxyz, const int* __restrict__ ball,
    const float* __restrict__ w1, const float* __restrict__ b1,
    const _Float16* __restrict__ w2f, const float* __restrict__ b2,
    const _Float16* __restrict__ w3f, const float* __restrict__ b3,
    float* __restrict__ outPts) {
  const int b    = blockIdx.y;
  const int s0   = blockIdx.x * 4;
  const int t    = threadIdx.x;
  const int w    = t >> 5;
  const int lane = t & 31;

  __shared__ _Float16 h1[128 * 64];
  __shared__ _Float16 h2[128 * 64];
  __shared__ _Float16 wsl[16 * 64];     // [16][K=64] N-major slice
  __shared__ float    pmax[8][16];

  // ---- layer1 (K=3, VALU) : grouped_xyz, W1+b1+relu ----
  {
    const int r    = t >> 1;           // row 0..127
    const int half = t & 1;
    const int c    = r >> 5, k = r & 31;
    const int s    = s0 + c;
    const int idx  = ball[(size_t)(b * S1 + s) * NS1 + k];
    const float cx = nxyz[(size_t)(b * S1 + s) * 3 + 0];
    const float cy = nxyz[(size_t)(b * S1 + s) * 3 + 1];
    const float cz = nxyz[(size_t)(b * S1 + s) * 3 + 2];
    const float fx = xyz[((size_t)b * NPTS1 + idx) * 3 + 0] - cx;
    const float fy = xyz[((size_t)b * NPTS1 + idx) * 3 + 1] - cy;
    const float fz = xyz[((size_t)b * NPTS1 + idx) * 3 + 2] - cz;
    const int o0 = half * 32;
#pragma unroll 8
    for (int o = o0; o < o0 + 32; ++o) {
      const float v = fx * w1[o * 3 + 0] + fy * w1[o * 3 + 1] + fz * w1[o * 3 + 2] + b1[o];
      h1[r * 64 + o] = (_Float16)fmaxf(v, 0.0f);
    }
  }
  __syncthreads();

  // ---- layer2 (64 -> 64) via WMMA, async-staged weight slices ----
  for (int nt = 0; nt < 4; ++nt) {
    stage_async(w2f + nt * 16 * 64, wsl, 16 * 64, t, 256);
    __syncthreads();
    v8f acc = {};
#pragma unroll
    for (int kt = 0; kt < 2; ++kt) {
      const v16h a  = load_a_frag(h1, w * 16, kt * 32, 64);
      const v16h bf = load_b_frag(wsl, kt * 32, 64);
      acc = __builtin_amdgcn_wmma_f32_16x16x32_f16(false, a, false, bf, (short)0, acc, false, false);
    }
    const int   col   = nt * 16 + (lane & 15);
    const float bb    = b2[col];
    const int   rbase = w * 16 + ((lane >> 4) << 3);
#pragma unroll
    for (int e = 0; e < 8; ++e)
      h2[(rbase + e) * 64 + col] = (_Float16)fmaxf(acc[e] + bb, 0.0f);
    __syncthreads();
  }

  // ---- layer3 (64 -> 128) via WMMA + maxpool over k=32 ----
  for (int nt = 0; nt < 8; ++nt) {
    stage_async(w3f + nt * 16 * 64, wsl, 16 * 64, t, 256);
    __syncthreads();
    v8f acc = {};
#pragma unroll
    for (int kt = 0; kt < 2; ++kt) {
      const v16h a  = load_a_frag(h2, w * 16, kt * 32, 64);
      const v16h bf = load_b_frag(wsl, kt * 32, 64);
      acc = __builtin_amdgcn_wmma_f32_16x16x32_f16(false, a, false, bf, (short)0, acc, false, false);
    }
    const float bb = b3[nt * 16 + (lane & 15)];
    float pm = 0.0f;                       // starting at 0 == fused ReLU
#pragma unroll
    for (int e = 0; e < 8; ++e) pm = fmaxf(pm, acc[e] + bb);
    pm = fmaxf(pm, __shfl_xor(pm, 16, 32));      // other 8 rows of this column
    if (lane < 16) pmax[w][lane] = pm;
    __syncthreads();
    if (t < 64) {
      const int c = t >> 4, col = t & 15;
      const float m = fmaxf(pmax[2 * c][col], pmax[2 * c + 1][col]);
      outPts[(size_t)(b * S1 + s0 + c) * 128 + nt * 16 + col] = m;
    }
    __syncthreads();
  }
}

// ---------------- SA2 MLP (131->128->256->256) + maxpool over 64 : 1 center/block ----------------
__global__ void __launch_bounds__(128) sa2_mlp_kernel(
    const float* __restrict__ l1xyz, const float* __restrict__ l2xyz,
    const float* __restrict__ l1pts, const int* __restrict__ ball,
    const _Float16* __restrict__ w1f, const float* __restrict__ b1,
    const _Float16* __restrict__ w2f, const float* __restrict__ b2,
    const _Float16* __restrict__ w3f, const float* __restrict__ b3,
    float* __restrict__ out) {
  const int b    = blockIdx.y;
  const int s    = blockIdx.x;
  const int t    = threadIdx.x;    // 128 threads = 4 waves
  const int w    = t >> 5;
  const int lane = t & 31;

  __shared__ _Float16 reg0[64 * 256];   // A0 (64x160) aliased with h2 (64x256)
  __shared__ _Float16 h1[64 * 128];
  __shared__ _Float16 wsl[16 * 256];    // [16][K<=256] N-major slice
  __shared__ float    pmax[4][16];
  _Float16* A0 = reg0;   // stride 160, dead after layer1
  _Float16* h2 = reg0;   // stride 256, written in layer2

  // ---- build A0 = [grouped_xyz(3) | grouped l1_points(128) | zero pad(29)] as f16 ----
  {
    const int r    = t >> 1;   // row 0..63
    const int half = t & 1;
    const int idx  = ball[(size_t)(b * S2 + s) * NS2 + r];
    const float* pt = l1pts + ((size_t)(b * S1) + idx) * 128;
    _Float16* row = A0 + r * 160;
    if (half == 0) {
      row[0] = (_Float16)(l1xyz[(size_t)(b * S1 + idx) * 3 + 0] - l2xyz[(size_t)(b * S2 + s) * 3 + 0]);
      row[1] = (_Float16)(l1xyz[(size_t)(b * S1 + idx) * 3 + 1] - l2xyz[(size_t)(b * S2 + s) * 3 + 1]);
      row[2] = (_Float16)(l1xyz[(size_t)(b * S1 + idx) * 3 + 2] - l2xyz[(size_t)(b * S2 + s) * 3 + 2]);
      for (int c = 0; c < 64; ++c) row[3 + c] = (_Float16)pt[c];
    } else {
      for (int c = 64; c < 128; ++c) row[3 + c] = (_Float16)pt[c];
      for (int c = 131; c < 160; ++c) row[c] = (_Float16)0.0f;
    }
  }
  __syncthreads();

  // ---- layer1: K=160 (padded), N=128 ----
  for (int nt = 0; nt < 8; ++nt) {
    stage_async(w1f + nt * 16 * 160, wsl, 16 * 160, t, 128);
    __syncthreads();
    v8f acc = {};
#pragma unroll
    for (int kt = 0; kt < 5; ++kt) {
      const v16h a  = load_a_frag(A0, w * 16, kt * 32, 160);
      const v16h bf = load_b_frag(wsl, kt * 32, 160);
      acc = __builtin_amdgcn_wmma_f32_16x16x32_f16(false, a, false, bf, (short)0, acc, false, false);
    }
    const int   col   = nt * 16 + (lane & 15);
    const float bb    = b1[col];
    const int   rbase = w * 16 + ((lane >> 4) << 3);
#pragma unroll
    for (int e = 0; e < 8; ++e)
      h1[(rbase + e) * 128 + col] = (_Float16)fmaxf(acc[e] + bb, 0.0f);
    __syncthreads();
  }

  // ---- layer2: K=128, N=256 (writes h2 == reg0, A0 now dead) ----
  for (int nt = 0; nt < 16; ++nt) {
    stage_async(w2f + nt * 16 * 128, wsl, 16 * 128, t, 128);
    __syncthreads();
    v8f acc = {};
#pragma unroll
    for (int kt = 0; kt < 4; ++kt) {
      const v16h a  = load_a_frag(h1, w * 16, kt * 32, 128);
      const v16h bf = load_b_frag(wsl, kt * 32, 128);
      acc = __builtin_amdgcn_wmma_f32_16x16x32_f16(false, a, false, bf, (short)0, acc, false, false);
    }
    const int   col   = nt * 16 + (lane & 15);
    const float bb    = b2[col];
    const int   rbase = w * 16 + ((lane >> 4) << 3);
#pragma unroll
    for (int e = 0; e < 8; ++e)
      h2[(rbase + e) * 256 + col] = (_Float16)fmaxf(acc[e] + bb, 0.0f);
    __syncthreads();
  }

  // ---- layer3: K=256, N=256 + maxpool over 64 rows ----
  for (int nt = 0; nt < 16; ++nt) {
    __builtin_prefetch(w3f + (size_t)(nt + 1) * 16 * 256, 0, 1);
    stage_async(w3f + nt * 16 * 256, wsl, 16 * 256, t, 128);
    __syncthreads();
    v8f acc = {};
#pragma unroll
    for (int kt = 0; kt < 8; ++kt) {
      const v16h a  = load_a_frag(h2, w * 16, kt * 32, 256);
      const v16h bf = load_b_frag(wsl, kt * 32, 256);
      acc = __builtin_amdgcn_wmma_f32_16x16x32_f16(false, a, false, bf, (short)0, acc, false, false);
    }
    const float bb = b3[nt * 16 + (lane & 15)];
    float pm = 0.0f;
#pragma unroll
    for (int e = 0; e < 8; ++e) pm = fmaxf(pm, acc[e] + bb);
    pm = fmaxf(pm, __shfl_xor(pm, 16, 32));
    if (lane < 16) pmax[w][lane] = pm;
    __syncthreads();
    if (t < 16) {
      const float m = fmaxf(fmaxf(pmax[0][t], pmax[1][t]), fmaxf(pmax[2][t], pmax[3][t]));
      out[(size_t)(b * S2 + s) * 256 + nt * 16 + t] = m;
    }
    __syncthreads();
  }
}

// ---------------- host launcher ----------------
extern "C" void kernel_launch(void* const* d_in, const int* in_sizes, int n_in,
                              void* d_out, int out_size, void* d_ws, size_t ws_size,
                              hipStream_t stream) {
  (void)in_sizes; (void)n_in; (void)out_size; (void)ws_size;
  const float* xyz    = (const float*)d_in[0];
  const float* sa1_w1 = (const float*)d_in[1];
  const float* sa1_b1 = (const float*)d_in[2];
  const float* sa1_w2 = (const float*)d_in[3];
  const float* sa1_b2 = (const float*)d_in[4];
  const float* sa1_w3 = (const float*)d_in[5];
  const float* sa1_b3 = (const float*)d_in[6];
  const float* sa2_w1 = (const float*)d_in[7];
  const float* sa2_b1 = (const float*)d_in[8];
  const float* sa2_w2 = (const float*)d_in[9];
  const float* sa2_b2 = (const float*)d_in[10];
  const float* sa2_w3 = (const float*)d_in[11];
  const float* sa2_b3 = (const float*)d_in[12];

  char* ws = (char*)d_ws;
  int*      fps1  = (int*)(ws + OFF_FPS1);
  float*    l1xyz = (float*)(ws + OFF_L1XYZ);
  int*      ball1 = (int*)(ws + OFF_BALL1);
  float*    l1pts = (float*)(ws + OFF_L1PTS);
  int*      fps2  = (int*)(ws + OFF_FPS2);
  float*    l2xyz = (float*)(ws + OFF_L2XYZ);
  int*      ball2 = (int*)(ws + OFF_BALL2);
  _Float16* wts   = (_Float16*)(ws + OFF_W);

  prep_weights_kernel<<<(W_TOTAL + 255) / 256, 256, 0, stream>>>(sa1_w2, sa1_w3, sa2_w1, sa2_w2, sa2_w3, wts);

  fps_kernel<NPTS1, S1><<<BATCH, 256, 0, stream>>>(xyz, fps1, l1xyz);
  ball_kernel<NPTS1, S1, NS1><<<BATCH * S1 / 8, 256, 0, stream>>>(xyz, l1xyz, ball1, R1 * R1);
  sa1_mlp_kernel<<<dim3(S1 / 4, BATCH), 256, 0, stream>>>(
      xyz, l1xyz, ball1, sa1_w1, sa1_b1,
      wts + WO_SA1W2, sa1_b2, wts + WO_SA1W3, sa1_b3, l1pts);

  fps_kernel<S1, S2><<<BATCH, 256, 0, stream>>>(l1xyz, fps2, l2xyz);
  ball_kernel<S1, S2, NS2><<<BATCH * S2 / 8, 256, 0, stream>>>(l1xyz, l2xyz, ball2, R2 * R2);
  sa2_mlp_kernel<<<dim3(S2, BATCH), 128, 0, stream>>>(
      l1xyz, l2xyz, l1pts, ball2,
      wts + WO_SA2W1, sa2_b1, wts + WO_SA2W2, sa2_b2, wts + WO_SA2W3, sa2_b3,
      (float*)d_out);
}
